// UnpoolWithSkip_69380901699941
// MI455X (gfx1250) — matrix operand order repaired and checked by reading, not compiled, not run on hardware
//
#include <hip/hip_runtime.h>
#include <hip/hip_bf16.h>

typedef __attribute__((ext_vector_type(16))) __bf16 v16bf;
typedef __attribute__((ext_vector_type(8)))  __bf16 v8bf;
typedef __attribute__((ext_vector_type(8)))  float  v8f;

#define BN_EPS 1e-5f

static __device__ __forceinline__ __bf16 f2bf(float f) {
  unsigned u = __builtin_bit_cast(unsigned, f);
  unsigned r = (u + 0x7FFFu + ((u >> 16) & 1u)) >> 16;
  unsigned short s = (unsigned short)r;
  return __builtin_bit_cast(__bf16, s);
}
static __device__ __forceinline__ float bf2f(__bf16 b) {
  unsigned u = ((unsigned)__builtin_bit_cast(unsigned short, b)) << 16;
  return __builtin_bit_cast(float, u);
}

static __device__ __forceinline__ v8f wmma_bf16(v16bf a, v16bf b, v8f c) {
  return __builtin_amdgcn_wmma_f32_16x16x32_bf16(false, a, false, b, (short)0, c,
                                                 false, false);
}

// ---- B-fragment staging: W[K,ldN] cols [t*16,t*16+16) -> LDS in 16x16x32 bf16
// B layout: per k-chunk q, lane half h holds K=q*32+8h+0..7 (elems 0..7) and
// K=q*32+16+8h+0..7 (elems 8..15).
static __device__ __forceinline__ void stage_B_tile(const float* W, int K, int ldN,
                                                    int t, __bf16* lds, int tid,
                                                    int nth) {
  int total = K * 16;
  for (int e = tid; e < total; e += nth) {
    int k = e >> 4, nn = e & 15;
    int q = k >> 5, kk = k & 31;
    int addr = ((q * 32 + ((kk & 8) << 1) + nn) << 4) + (kk & 7) + ((kk & 16) >> 1);
    lds[addr] = f2bf(W[(size_t)k * ldN + t * 16 + nn]);
  }
}
static __device__ __forceinline__ v16bf load_B(const __bf16* lds, int q, int lane) {
  return *(const v16bf*)(lds + ((q * 32 + lane) << 4));
}

// A fragment from row-major bf16: lane row = lane&15; elems 0..7 = K q*32+h8+0..7,
// elems 8..15 = K q*32+h8+16..23, h8 = 8*(lane>>4).  Two 16B loads, no VALU.
static __device__ __forceinline__ v16bf load_A_bf(const __bf16* A, int ld, int row0,
                                                  int q, int lane) {
  const __bf16* src =
      A + (size_t)(row0 + (lane & 15)) * ld + (q << 5) + ((lane >> 4) << 3);
  v8bf lo = *(const v8bf*)(src);
  v8bf hi = *(const v8bf*)(src + 16);
  return __builtin_shufflevector(lo, hi, 0, 1, 2, 3, 4, 5, 6, 7, 8, 9, 10, 11, 12,
                                 13, 14, 15);
}

// =====================================================================
// Generic GEMM: out[Mrows,Ncols](f32) = A16[Mrows,K](bf16) @ W[K,Ncols](f32)
// (+bias), optional fused per-channel sum / sumsq for training-mode BN.
// grid.x = Ncols/16 col tile, grid.y*8 waves stride row tiles.
// =====================================================================
__global__ void gemm_wmma(const __bf16* __restrict__ A, const float* __restrict__ W,
                          const float* __restrict__ bias, float* __restrict__ out,
                          float* __restrict__ ssum, float* __restrict__ ssqs,
                          int Mrows, int K, int Ncols) {
  __shared__ __attribute__((aligned(32))) __bf16 ldsB[256 * 16];  // up to K=256
  int t = blockIdx.x;
  stage_B_tile(W, K, Ncols, t, ldsB, threadIdx.x, blockDim.x);
  __syncthreads();
  int lane = threadIdx.x & 31, wave = threadIdx.x >> 5;
  int nq = K >> 5;
  int rowTiles = Mrows >> 4;
  int stride = gridDim.y * 8;
  int col = (t << 4) + (lane & 15);
  int h8 = (lane >> 4) << 3;
  float bv = bias ? bias[col] : 0.f;
  float aS = 0.f, aQ = 0.f;
  for (int rt = blockIdx.y * 8 + wave; rt < rowTiles; rt += stride) {
    v8f acc = {};
    for (int q = 0; q < nq; ++q) {
      v16bf a = load_A_bf(A, K, rt << 4, q, lane);
      v16bf b = load_B(ldsB, q, lane);
      acc = wmma_bf16(a, b, acc);
    }
#pragma unroll
    for (int r = 0; r < 8; ++r) {
      float v = acc[r] + bv;
      out[(size_t)((rt << 4) + r + h8) * Ncols + col] = v;
      aS += v;
      aQ += v * v;
    }
  }
  if (ssum) {
    aS += __shfl_xor(aS, 16, 32);
    aQ += __shfl_xor(aQ, 16, 32);
    if (lane < 16) {
      atomicAdd(&ssum[col], aS);
      atomicAdd(&ssqs[col], aQ);
    }
  }
}

__global__ void f32_to_bf16_k(const float* __restrict__ x, __bf16* __restrict__ y,
                              size_t total) {
  size_t i = (size_t)blockIdx.x * blockDim.x + threadIdx.x;
  size_t stride = (size_t)gridDim.x * blockDim.x;
  for (; i < total; i += stride) y[i] = f2bf(x[i]);
}

// stat[c]=sum, stat[C+c]=sumsq  ->  sc[c]=scale, sc[C+c]=shift
__global__ void finalize_bn(const float* __restrict__ stat,
                            const float* __restrict__ gamma,
                            const float* __restrict__ beta, float* __restrict__ sc,
                            float cnt, int C) {
  int c = blockIdx.x * blockDim.x + threadIdx.x;
  if (c < C) {
    float mean = stat[c] / cnt;
    float var = stat[C + c] / cnt - mean * mean;
    float s = gamma[c] * rsqrtf(var + BN_EPS);
    sc[c] = s;
    sc[C + c] = beta[c] - mean * s;
  }
}

// x = relu(bn(x)) in place (f32), plus optional bf16 mirror for GEMM A-operands.
__global__ void bn_relu_bf(float* __restrict__ x, __bf16* __restrict__ y,
                           const float* __restrict__ sc, size_t total) {
  size_t i = (size_t)blockIdx.x * blockDim.x + threadIdx.x;
  size_t stride = (size_t)gridDim.x * blockDim.x;
  for (; i < total; i += stride) {
    int c = (int)(i & 255);  // C = 256
    float v = fmaxf(x[i] * sc[c] + sc[256 + c], 0.f);
    x[i] = v;
    if (y) y[i] = f2bf(v);
  }
}

// hres (query-GEMM input only) produced directly in bf16.
__global__ void hres16_kernel(const float* __restrict__ lres,
                              const float* __restrict__ skip,
                              const int* __restrict__ cluster,
                              __bf16* __restrict__ hres16, int M) {
  size_t total = (size_t)M * 256;
  size_t i = (size_t)blockIdx.x * blockDim.x + threadIdx.x;
  size_t stride = (size_t)gridDim.x * blockDim.x;
  for (; i < total; i += stride) {
    int m = (int)(i >> 8), c = (int)(i & 255);
    hres16[i] = f2bf(lres[(size_t)cluster[m] * 256 + c] + skip[i]);
  }
}

// pos = coord[ref[m,s]] - skip_coord[m]; accumulate Sum(pos), Sum(pos x pos).
__global__ void pos_stats(const float* __restrict__ coord,
                          const float* __restrict__ skipc,
                          const int* __restrict__ cluster,
                          const int* __restrict__ refidx, float* __restrict__ stat,
                          int M) {
  int total = M * 16;
  int i = blockIdx.x * blockDim.x + threadIdx.x;
  int stride = gridDim.x * blockDim.x;
  float a[9];
#pragma unroll
  for (int j = 0; j < 9; ++j) a[j] = 0.f;
  for (; i < total; i += stride) {
    int m = i >> 4, s = i & 15;
    int r = refidx[cluster[m] * 16 + s];
    float x = coord[(size_t)r * 3 + 0] - skipc[(size_t)m * 3 + 0];
    float y = coord[(size_t)r * 3 + 1] - skipc[(size_t)m * 3 + 1];
    float z = coord[(size_t)r * 3 + 2] - skipc[(size_t)m * 3 + 2];
    a[0] += x; a[1] += y; a[2] += z;
    a[3] += x * x; a[4] += x * y; a[5] += x * z;
    a[6] += y * y; a[7] += y * z; a[8] += z * z;
  }
#pragma unroll
  for (int off = 16; off; off >>= 1)
#pragma unroll
    for (int j = 0; j < 9; ++j) a[j] += __shfl_xor(a[j], off, 32);
  if ((threadIdx.x & 31) == 0)
#pragma unroll
    for (int j = 0; j < 9; ++j) atomicAdd(&stat[j], a[j]);
}

// Analytic BN params for bn(pos@Wp1+bp1) from 3-dim mean + 3x3 covariance.
__global__ void finalize_pos(const float* __restrict__ stat,
                             const float* __restrict__ Wp1,
                             const float* __restrict__ bp1,
                             const float* __restrict__ gp1,
                             const float* __restrict__ bep1, float* __restrict__ s1c1,
                             float cnt) {
  int c = threadIdx.x;
  float m0 = stat[0] / cnt, m1 = stat[1] / cnt, m2 = stat[2] / cnt;
  float c00 = stat[3] / cnt - m0 * m0;
  float c01 = stat[4] / cnt - m0 * m1;
  float c02 = stat[5] / cnt - m0 * m2;
  float c11 = stat[6] / cnt - m1 * m1;
  float c12 = stat[7] / cnt - m1 * m2;
  float c22 = stat[8] / cnt - m2 * m2;
  float w0 = Wp1[c], w1 = Wp1[256 + c], w2 = Wp1[512 + c];
  float mu = bp1[c] + m0 * w0 + m1 * w1 + m2 * w2;
  float var = w0 * w0 * c00 + w1 * w1 * c11 + w2 * w2 * c22 +
              2.f * (w0 * w1 * c01 + w0 * w2 * c02 + w1 * w2 * c12);
  float s = gp1[c] * rsqrtf(var + BN_EPS);
  s1c1[c] = s;
  s1c1[256 + c] = (bp1[c] - mu) * s + bep1[c];
}

// Wpw = Wp2 @ Ww1 [256,16]; cw[g] = bp2 @ Ww1 + bw1
__global__ void fuse_w(const float* __restrict__ Wp2, const float* __restrict__ Ww1,
                       const float* __restrict__ bp2, const float* __restrict__ bw1,
                       float* __restrict__ Wpw, float* __restrict__ cw) {
  int j = threadIdx.x;
  for (int g = 0; g < 16; ++g) {
    float acc = 0.f;
    for (int c = 0; c < 256; ++c) acc += Wp2[(size_t)j * 256 + c] * Ww1[c * 16 + g];
    Wpw[j * 16 + g] = acc;
  }
  if (j < 16) {
    float acc = bw1[j];
    for (int c = 0; c < 256; ++c) acc += bp2[c] * Ww1[c * 16 + j];
    cw[j] = acc;
  }
}

// logits[m,s,g] = H@(Wp2@Ww1) + key_w[ref] - query_w[m] + cw (pre-BN2), fused
// channel stats for BN2, AND materializes H (bf16, row-major, WMMA-A-friendly)
// so the heavy out_kernel never recomputes it.
__global__ void logits_kernel(const int* __restrict__ cluster,
                              const int* __restrict__ refidx,
                              const float* __restrict__ coord,
                              const float* __restrict__ skipc,
                              const float* __restrict__ Wp1,
                              const float* __restrict__ s1c1,
                              const float* __restrict__ Wpw,
                              const float* __restrict__ cw,
                              const float* __restrict__ keyw,
                              const float* __restrict__ queryw,
                              __bf16* __restrict__ Hbuf,
                              float* __restrict__ logits, float* __restrict__ stat2,
                              int M) {
  __shared__ __attribute__((aligned(32))) __bf16 ldsB[256 * 16];
  __shared__ float wp1s[768], s1s[256], c1s[256];
  stage_B_tile(Wpw, 256, 16, 0, ldsB, threadIdx.x, blockDim.x);
  for (int e = threadIdx.x; e < 768; e += blockDim.x) wp1s[e] = Wp1[e];
  for (int e = threadIdx.x; e < 256; e += blockDim.x) {
    s1s[e] = s1c1[e];
    c1s[e] = s1c1[256 + e];
  }
  __syncthreads();
  int lane = threadIdx.x & 31, wave = threadIdx.x >> 5;
  int wg = blockIdx.x * 8 + wave;
  int g = lane & 15, s = lane & 15, h8 = (lane >> 4) << 3;
  float aS = 0.f, aQ = 0.f;
  for (int i = 0; i < 8; ++i) {
    int m = wg * 8 + i;
    if (m >= M) break;
    int cl = cluster[m];
    int refr[8];
#pragma unroll
    for (int r = 0; r < 8; ++r) refr[r] = refidx[cl * 16 + r + h8];
    int rs = refidx[cl * 16 + s];
    float p0 = coord[(size_t)rs * 3 + 0] - skipc[(size_t)m * 3 + 0];
    float p1 = coord[(size_t)rs * 3 + 1] - skipc[(size_t)m * 3 + 1];
    float p2 = coord[(size_t)rs * 3 + 2] - skipc[(size_t)m * 3 + 2];
    v16bf A[8];
#pragma unroll
    for (int q = 0; q < 8; ++q) {
#pragma unroll
      for (int j = 0; j < 16; ++j) {
        int k = (q << 5) + h8 + ((j < 8) ? j : (j + 8));
        float t = p0 * wp1s[k] + p1 * wp1s[256 + k] + p2 * wp1s[512 + k];
        A[q][j] = f2bf(fmaxf(t * s1s[k] + c1s[k], 0.f));
      }
      // persist this lane's two 8-element runs of H row s (row-major bf16)
      __bf16* hd = Hbuf + (((size_t)m * 16 + s) << 8) + (q << 5) + h8;
      *(v8bf*)hd = __builtin_shufflevector(A[q], A[q], 0, 1, 2, 3, 4, 5, 6, 7);
      *(v8bf*)(hd + 16) =
          __builtin_shufflevector(A[q], A[q], 8, 9, 10, 11, 12, 13, 14, 15);
    }
    v8f acc = {};
#pragma unroll
    for (int q = 0; q < 8; ++q) acc = wmma_bf16(A[q], load_B(ldsB, q, lane), acc);
    float qw = queryw[(size_t)m * 16 + g];
    float cwg = cw[g];
#pragma unroll
    for (int r = 0; r < 8; ++r) {
      int row = r + h8;
      float lg = acc[r] + keyw[(size_t)refr[r] * 16 + g] - qw + cwg;
      logits[(((size_t)m * 16 + row) << 4) + g] = lg;
      aS += lg;
      aQ += lg * lg;
    }
  }
  aS += __shfl_xor(aS, 16, 32);
  aQ += __shfl_xor(aQ, 16, 32);
  if (lane < 16) {
    atomicAdd(&stat2[lane], aS);
    atomicAdd(&stat2[16 + lane], aQ);
  }
}

// weight = softmax_s( relu(bn2(logits)) @ Ww2 + bw2 ) * mask, in place.
__global__ void weights_kernel(float* __restrict__ logw,
                               const int* __restrict__ cluster,
                               const int* __restrict__ refidx,
                               const float* __restrict__ sc2,
                               const float* __restrict__ Ww2,
                               const float* __restrict__ bw2, int M) {
  __shared__ float ybuf[16][16][16];
  __shared__ float maskb[16][16];
  int mi = threadIdx.x >> 4, si = threadIdx.x & 15;
  int m = blockIdx.x * 16 + mi;
  int cl = cluster[m];
  int r = refidx[cl * 16 + si];
  int rp1 = r + 1;
  maskb[mi][si] = (rp1 > 0) ? 1.f : ((rp1 < 0) ? -1.f : 0.f);
  const float* lp = logw + (((size_t)m * 16 + si) << 4);
  float rr[16];
#pragma unroll
  for (int g = 0; g < 16; ++g) rr[g] = fmaxf(lp[g] * sc2[g] + sc2[16 + g], 0.f);
#pragma unroll
  for (int go = 0; go < 16; ++go) {
    float acc = bw2[go];
#pragma unroll
    for (int g = 0; g < 16; ++g) acc += rr[g] * Ww2[g * 16 + go];
    ybuf[mi][si][go] = acc;
  }
  __syncthreads();
  float mx = -3.0e38f;
#pragma unroll
  for (int s2 = 0; s2 < 16; ++s2) mx = fmaxf(mx, ybuf[mi][s2][si]);
  float e[16], sum = 0.f;
#pragma unroll
  for (int s2 = 0; s2 < 16; ++s2) {
    e[s2] = __expf(ybuf[mi][s2][si] - mx);
    sum += e[s2];
  }
  float inv = 1.f / sum;
#pragma unroll
  for (int s2 = 0; s2 < 16; ++s2)
    logw[(((size_t)m * 16 + s2) << 4) + si] = e[s2] * inv * maskb[mi][s2];
}

// out[m,c] = sum_s (lres16[ref]+H@Wp2+bp2)*weight[m,s,c/16] + skip_p[m,c].
// A-fragments come straight from the bf16 H buffer (2x b128 per chunk).
// grid.x = 4 tile-quads (adjacent blocks share H rows in L2), grid.y = m-groups.
__global__ void out_kernel(const int* __restrict__ cluster,
                           const int* __restrict__ refidx,
                           const __bf16* __restrict__ Hbuf,
                           const float* __restrict__ Wp2,
                           const float* __restrict__ bp2,
                           const __bf16* __restrict__ lres16,
                           const float* __restrict__ skip_p,
                           const float* __restrict__ weight, float* __restrict__ out,
                           int M) {
  __shared__ __attribute__((aligned(32))) __bf16 ldsB[4 * 256 * 16];  // 32KB
  int t0 = blockIdx.x * 4;
#pragma unroll
  for (int tt = 0; tt < 4; ++tt)
    stage_B_tile(Wp2, 256, 256, t0 + tt, ldsB + tt * 4096, threadIdx.x, blockDim.x);
  __syncthreads();
  int lane = threadIdx.x & 31, wave = threadIdx.x >> 5;
  int wg = blockIdx.y * 8 + wave;
  int nn = lane & 15, h8 = (lane >> 4) << 3;
  for (int i = 0; i < 8; ++i) {
    int m = wg * 8 + i;
    if (m >= M) break;
    int cl = cluster[m];
    int refr[8];
#pragma unroll
    for (int r = 0; r < 8; ++r) refr[r] = refidx[cl * 16 + r + h8];
    v16bf A[8];
#pragma unroll
    for (int q = 0; q < 8; ++q) A[q] = load_A_bf(Hbuf, 256, m * 16, q, lane);
#pragma unroll
    for (int tt = 0; tt < 4; ++tt) {
      int t = t0 + tt;
      v8f acc = {};
#pragma unroll
      for (int q = 0; q < 8; ++q)
        acc = wmma_bf16(A[q], load_B(ldsB + tt * 4096, q, lane), acc);
      float part = 0.f, ws = 0.f;
#pragma unroll
      for (int r = 0; r < 8; ++r) {
        int row = r + h8;
        float w = weight[(((size_t)m * 16 + row) << 4) + t];
        float v = bf2f(lres16[(size_t)refr[r] * 256 + (t << 4) + nn]);
        part += (acc[r] + v) * w;
        ws += w;
      }
      part += __shfl_xor(part, 16, 32);
      ws += __shfl_xor(ws, 16, 32);
      if (lane < 16) {
        int c = (t << 4) + lane;
        out[(size_t)m * 256 + c] = part + bp2[c] * ws + skip_p[(size_t)m * 256 + c];
      }
    }
  }
}

extern "C" void kernel_launch(void* const* d_in, const int* in_sizes, int n_in,
                              void* d_out, int out_size, void* d_ws, size_t ws_size,
                              hipStream_t stream) {
  const float* coord = (const float*)d_in[0];
  const float* feat  = (const float*)d_in[1];
  const float* skipc = (const float*)d_in[2];
  const float* skipf = (const float*)d_in[3];
  const int* cluster = (const int*)d_in[4];
  const int* refidx  = (const int*)d_in[5];
  const float *Wp = (const float*)d_in[6],  *bp = (const float*)d_in[7];
  const float *gp = (const float*)d_in[8],  *bep = (const float*)d_in[9];
  const float *Ws = (const float*)d_in[10], *bs = (const float*)d_in[11];
  const float *gs = (const float*)d_in[12], *bes = (const float*)d_in[13];
  const float *Wk = (const float*)d_in[14], *bk = (const float*)d_in[15];
  const float *gk = (const float*)d_in[16], *bek = (const float*)d_in[17];
  const float *Wq = (const float*)d_in[18], *bq = (const float*)d_in[19];
  const float *gq = (const float*)d_in[20], *beq = (const float*)d_in[21];
  const float *Wp1 = (const float*)d_in[22], *bp1 = (const float*)d_in[23];
  const float *gp1 = (const float*)d_in[24], *bep1 = (const float*)d_in[25];
  const float *Wp2 = (const float*)d_in[26], *bp2 = (const float*)d_in[27];
  const float *Ww1 = (const float*)d_in[28], *bw1 = (const float*)d_in[29];
  const float *gw1 = (const float*)d_in[30], *bew1 = (const float*)d_in[31];
  const float *Ww2 = (const float*)d_in[32], *bw2 = (const float*)d_in[33];

  const int N = in_sizes[0] / 3;   // 16384
  const int M = in_sizes[2] / 3;   // 65536
  const int Cin = in_sizes[1] / N; // 256
  const int Csk = in_sizes[3] / M; // 128

  float* wsf = (float*)d_ws;
  size_t o = 0;
  auto alloc = [&](size_t n) { size_t r = o; o += (n + 63) & ~(size_t)63; return r; };
  // f32 buffers
  size_t o_lres = alloc((size_t)N * 256);
  size_t o_skip = alloc((size_t)M * 256);
  size_t o_key  = alloc((size_t)N * 256);
  size_t o_qry  = alloc((size_t)M * 256);
  size_t o_keyw = alloc((size_t)N * 16);
  size_t o_qryw = alloc((size_t)M * 16);
  size_t o_logw = alloc((size_t)M * 256);
  // bf16 buffers (sizes in float units = half the element count)
  size_t o_feat16 = alloc((size_t)N * (Cin / 2));
  size_t o_skpf16 = alloc((size_t)M * (Csk / 2));
  size_t o_lres16 = alloc((size_t)N * 128);
  size_t o_hres16 = alloc((size_t)M * 128);
  size_t o_key16  = alloc((size_t)N * 128);
  size_t o_qry16  = alloc((size_t)M * 128);
  size_t o_H      = alloc((size_t)M * 16 * 128);  // [M*16, 256] bf16
  // stats / params
  size_t o_statP = alloc(512);
  size_t o_statS = alloc(512);
  size_t o_statK = alloc(512);
  size_t o_statQ = alloc(512);
  size_t o_stat2 = alloc(64);
  size_t o_statPos = alloc(64);
  size_t statEnd = o;
  size_t o_scP = alloc(512);
  size_t o_scS = alloc(512);
  size_t o_scK = alloc(512);
  size_t o_scQ = alloc(512);
  size_t o_sc2 = alloc(64);
  size_t o_s1c1 = alloc(512);
  size_t o_Wpw = alloc(4096);
  size_t o_cw = alloc(64);

  hipMemsetAsync(wsf + o_statP, 0, (statEnd - o_statP) * sizeof(float), stream);

  auto rg = [](int rows) { int y = (rows >> 4) / 64; return y < 1 ? 1 : y; };
  __bf16* feat16 = (__bf16*)(wsf + o_feat16);
  __bf16* skpf16 = (__bf16*)(wsf + o_skpf16);
  __bf16* lres16 = (__bf16*)(wsf + o_lres16);
  __bf16* hres16 = (__bf16*)(wsf + o_hres16);
  __bf16* key16  = (__bf16*)(wsf + o_key16);
  __bf16* qry16  = (__bf16*)(wsf + o_qry16);
  __bf16* Hbuf   = (__bf16*)(wsf + o_H);

  // independent prep
  f32_to_bf16_k<<<2048, 256, 0, stream>>>(feat, feat16, (size_t)N * Cin);
  f32_to_bf16_k<<<4096, 256, 0, stream>>>(skipf, skpf16, (size_t)M * Csk);
  fuse_w<<<1, 256, 0, stream>>>(Wp2, Ww1, bp2, bw1, wsf + o_Wpw, wsf + o_cw);
  pos_stats<<<512, 256, 0, stream>>>(coord, skipc, cluster, refidx,
                                     wsf + o_statPos, M);
  finalize_pos<<<1, 256, 0, stream>>>(wsf + o_statPos, Wp1, bp1, gp1, bep1,
                                      wsf + o_s1c1, (float)((size_t)M * 16));

  // proj + proj_skip GEMMs with fused BN stats
  gemm_wmma<<<dim3(16, rg(N)), 256, 0, stream>>>(feat16, Wp, bp, wsf + o_lres,
                                                 wsf + o_statP, wsf + o_statP + 256,
                                                 N, Cin, 256);
  gemm_wmma<<<dim3(16, rg(M)), 256, 0, stream>>>(skpf16, Ws, bs, wsf + o_skip,
                                                 wsf + o_statS, wsf + o_statS + 256,
                                                 M, Csk, 256);
  finalize_bn<<<1, 256, 0, stream>>>(wsf + o_statP, gp, bep, wsf + o_scP, (float)N, 256);
  finalize_bn<<<1, 256, 0, stream>>>(wsf + o_statS, gs, bes, wsf + o_scS, (float)M, 256);
  bn_relu_bf<<<2048, 256, 0, stream>>>(wsf + o_lres, lres16, wsf + o_scP,
                                       (size_t)N * 256);
  bn_relu_bf<<<8192, 256, 0, stream>>>(wsf + o_skip, (__bf16*)nullptr, wsf + o_scS,
                                       (size_t)M * 256);

  hres16_kernel<<<8192, 256, 0, stream>>>(wsf + o_lres, wsf + o_skip, cluster,
                                          hres16, M);

  // key / query GEMMs + BN (+bf16 mirrors for the Ww1 projections)
  gemm_wmma<<<dim3(16, rg(N)), 256, 0, stream>>>(lres16, Wk, bk, wsf + o_key,
                                                 wsf + o_statK, wsf + o_statK + 256,
                                                 N, 256, 256);
  finalize_bn<<<1, 256, 0, stream>>>(wsf + o_statK, gk, bek, wsf + o_scK, (float)N, 256);
  bn_relu_bf<<<2048, 256, 0, stream>>>(wsf + o_key, key16, wsf + o_scK,
                                       (size_t)N * 256);

  gemm_wmma<<<dim3(16, rg(M)), 256, 0, stream>>>(hres16, Wq, bq, wsf + o_qry,
                                                 wsf + o_statQ, wsf + o_statQ + 256,
                                                 M, 256, 256);
  finalize_bn<<<1, 256, 0, stream>>>(wsf + o_statQ, gq, beq, wsf + o_scQ, (float)M, 256);
  bn_relu_bf<<<8192, 256, 0, stream>>>(wsf + o_qry, qry16, wsf + o_scQ,
                                       (size_t)M * 256);

  // factored logit path: key@Ww1, query@Ww1
  gemm_wmma<<<dim3(1, rg(N)), 256, 0, stream>>>(key16, Ww1, nullptr, wsf + o_keyw,
                                                nullptr, nullptr, N, 256, 16);
  gemm_wmma<<<dim3(1, rg(M)), 256, 0, stream>>>(qry16, Ww1, nullptr, wsf + o_qryw,
                                                nullptr, nullptr, M, 256, 16);

  logits_kernel<<<M / 64, 256, 0, stream>>>(cluster, refidx, coord, skipc, Wp1,
                                            wsf + o_s1c1, wsf + o_Wpw, wsf + o_cw,
                                            wsf + o_keyw, wsf + o_qryw, Hbuf,
                                            wsf + o_logw, wsf + o_stat2, M);
  finalize_bn<<<1, 256, 0, stream>>>(wsf + o_stat2, gw1, bew1, wsf + o_sc2,
                                     (float)((size_t)M * 16), 16);
  weights_kernel<<<M / 16, 256, 0, stream>>>(wsf + o_logw, cluster, refidx,
                                             wsf + o_sc2, Ww2, bw2, M);

  out_kernel<<<dim3(4, M / 64), 256, 0, stream>>>(cluster, refidx, Hbuf, Wp2, bp2,
                                                  lres16, wsf + o_skip,
                                                  wsf + o_logw, (float*)d_out, M);
}
